// PredRNN_85847806313212
// MI455X (gfx1250) — compile-verified, matrix operand order checked by reading, not run on hardware
//
#include <hip/hip_runtime.h>
#include <cmath>

// ---------------------------------------------------------------------------
// PredRNN (2x ConvLSTM F=64 + BN + 3D out-conv) for MI455X / gfx1250.
// Core: fused per-timestep implicit-GEMM ConvLSTM step using
// v_wmma_f32_16x16x32_f16 (wave32), M=131072 N=256 K=576(+input chunks).
// K loop: double-buffered LDS tiles, 1 barrier/chunk. Weight tiles are
// prefetched with GLOBAL_LOAD_ASYNC_TO_LDS_B128 (ASYNCcnt, zero data VGPRs)
// so the whole loop fits the CU-mode 128-VGPR budget without spills.
// ---------------------------------------------------------------------------

typedef __attribute__((ext_vector_type(16))) _Float16 v16h;
typedef __attribute__((ext_vector_type(8)))  float    v8f;

constexpr int BB = 8, TT = 10, HH = 128, WW = 128, CC = 3, FF = 64;
constexpr int HW    = HH * WW;        // 16384
constexpr int MTOT  = BB * HW;        // 131072 GEMM rows per timestep
constexpr int NGATE = 4 * FF;         // 256 gate channels (i,f,c,o)
constexpr int BLK_M = 64;             // M tile per workgroup
constexpr int NTH   = 256;            // 8 waves (wave32)
constexpr int A_STRIDE = 40;          // halfs per LDS A row (32 data + 8 pad -> conflict-free)
constexpr int B_STRIDE = 40;
constexpr int KCH_H = 18;             // K-chunks for a 3x3 conv over 64 ch (9 taps * 2)
constexpr int BUF_HALFS = BLK_M * A_STRIDE + NGATE * B_STRIDE;  // 25.6KB per buffer

__device__ __forceinline__ float hsig(float x) {
  return fminf(fmaxf(0.2f * x + 0.5f, 0.0f), 1.0f);
}

// ---------------------------------------------------------------------------
// Fused ConvLSTM step:  z = conv3x3(in, Wx) [+ conv3x3(h_prev, Wh)] + b
//                       gates -> c_new, h_new   (one 64x256 tile per WG)
// IN_C4:    input is 4-channel-padded x (2 K-chunks) vs 64-channel (18 chunks)
// HAS_STATE: t>0 (include Wh conv and c_prev)
// ---------------------------------------------------------------------------
template <bool IN_C4, bool HAS_STATE>
__global__ __launch_bounds__(NTH) void lstm_step_kernel(
    const _Float16* __restrict__ in,      // (B,HW,CIeff) slice for this t
    const _Float16* __restrict__ hprev,   // (B,HW,64) or unused
    const _Float16* __restrict__ wxp,     // packed [IN_CHUNKS][256][32]
    const _Float16* __restrict__ whp,     // packed [18][256][32]
    const float*    __restrict__ bias,    // 256
    _Float16*       __restrict__ hout,    // (B,HW,64)
    float*          __restrict__ cstate)  // (B,HW,64) read c_prev / write c_new
{
  constexpr int IN_CHUNKS = IN_C4 ? 2 : KCH_H;
  constexpr int TOT_CH    = IN_CHUNKS + (HAS_STATE ? KCH_H : 0);

  __shared__ __align__(16) unsigned char smem[65536];
  // double-buffered tiles: buf k at halfs offset k*BUF_HALFS (A first, then B)
  float* ldsC = reinterpret_cast<float*>(smem);  // 64x256 f32 epilogue exchange (aliased)

  const int tid  = threadIdx.x;
  const int lane = tid & 31;
  const int wave = tid >> 5;
  const int wm   = wave >> 2;       // M half (0..1) -> 32 rows
  const int wn   = wave & 3;        // N quarter (0..3) -> 64 cols
  const int hsel = lane >> 4;       // fragment half select
  const int l16  = lane & 15;

  const int gm0 = blockIdx.x * BLK_M;   // 64 consecutive rows share (b, y)
  const int bb  = gm0 / HW;
  const int yy  = (gm0 % HW) / WW;
  const int x0  = gm0 % WW;             // 0 or 64

  const int ar = tid >> 2;              // A-loader row (0..63)
  const int as = tid & 3;               // A-loader segment (8 halfs / 16B)

  auto bufA = [&](int k) -> _Float16* {
    return reinterpret_cast<_Float16*>(smem) + (size_t)k * BUF_HALFS;
  };
  auto bufB = [&](int k) -> _Float16* {
    return reinterpret_cast<_Float16*>(smem) + (size_t)k * BUF_HALFS + BLK_M * A_STRIDE;
  };
  // LDS byte offset of smem (flat LDS addressing: low 32 bits = LDS offset)
  const unsigned ldsBase = (unsigned)(uintptr_t)(void*)smem;
  // fragment base offsets (halfs) inside a buffer -- invariant across chunks
  const int aFragOff0 = (wm * 32 + l16) * A_STRIDE + hsel * 8;          // mf=0
  const int aFragOff1 = (wm * 32 + 16 + l16) * A_STRIDE + hsel * 8;     // mf=1
  const int bFragBase = (wn * 64 + l16) * B_STRIDE + hsel * 8;          // nf adds nf*16*B_STRIDE

  // ---- A stager: global -> register (im2col + zero padding on the fly) ----
  auto loadA = [&](int ch) -> uint4 {
    uint4 r = {0u, 0u, 0u, 0u};
    if (IN_C4 && ch < IN_CHUNKS) {
      // x input: taps packed 4 channels each (ch3 pre-zeroed; weight pads zero)
      if (ch == 0) {
        uint2 lo = {0u, 0u}, hi = {0u, 0u};
        const int tap0 = as * 2, tap1 = as * 2 + 1;
        {
          const int py = yy + tap0 / 3 - 1, px = x0 + ar + tap0 % 3 - 1;
          if ((unsigned)py < (unsigned)HH && (unsigned)px < (unsigned)WW)
            lo = *reinterpret_cast<const uint2*>(in + (size_t)(bb * HW + py * WW + px) * 4);
        }
        {
          const int py = yy + tap1 / 3 - 1, px = x0 + ar + tap1 % 3 - 1;
          if ((unsigned)py < (unsigned)HH && (unsigned)px < (unsigned)WW)
            hi = *reinterpret_cast<const uint2*>(in + (size_t)(bb * HW + py * WW + px) * 4);
        }
        r.x = lo.x; r.y = lo.y; r.z = hi.x; r.w = hi.y;
      } else if (as == 0) {             // chunk 1: only tap 8 (dy=+1,dx=+1)
        const int py = yy + 1, px = x0 + ar + 1;
        if ((unsigned)py < (unsigned)HH && (unsigned)px < (unsigned)WW) {
          const uint2 lo =
              *reinterpret_cast<const uint2*>(in + (size_t)(bb * HW + py * WW + px) * 4);
          r.x = lo.x; r.y = lo.y;
        }
      }
    } else {
      // 64-channel source: chunk -> (tap, 32-channel half); contiguous 16B loads
      const _Float16* asrc = (ch < IN_CHUNKS) ? in : hprev;
      const int cc   = (ch < IN_CHUNKS) ? ch : (ch - IN_CHUNKS);
      const int tap  = cc >> 1;
      const int coff = (cc & 1) * 32 + as * 8;
      const int py   = yy + tap / 3 - 1;
      const int px   = x0 + ar + tap % 3 - 1;
      if ((unsigned)py < (unsigned)HH && (unsigned)px < (unsigned)WW)
        r = *reinterpret_cast<const uint4*>(asrc + (size_t)(bb * HW + py * WW + px) * FF + coff);
    }
    return r;
  };
  auto storeA = [&](int k, uint4 av) {
    *reinterpret_cast<uint4*>(bufA(k) + ar * A_STRIDE + as * 8) = av;
  };
  // ---- B stager: global -> LDS with async DMA (no data VGPRs, ASYNCcnt) ----
  auto asyncB = [&](int ch, int k) {
    const _Float16* bsrc = (ch < IN_CHUNKS)
                               ? (wxp + (size_t)ch * NGATE * 32)
                               : (whp + (size_t)(ch - IN_CHUNKS) * NGATE * 32);
    unsigned long long g = (unsigned long long)(uintptr_t)(bsrc + (size_t)tid * 32);
    unsigned l = ldsBase + (unsigned)(k * BUF_HALFS + BLK_M * A_STRIDE + tid * B_STRIDE) * 2u;
#pragma unroll
    for (int s = 0; s < 4; ++s) {
      asm volatile("global_load_async_to_lds_b128 %0, %1, off"
                   :: "v"(l + (unsigned)(s * 16)), "v"(g + (unsigned long long)(s * 16))
                   : "memory");
    }
  };

  const v8f vz = {0.f, 0.f, 0.f, 0.f, 0.f, 0.f, 0.f, 0.f};
  v8f acc[2][4];
#pragma unroll
  for (int i = 0; i < 2; ++i)
#pragma unroll
    for (int j = 0; j < 4; ++j) acc[i][j] = vz;

  // ---- pipeline prologue: stage chunk 0 ----
  storeA(0, loadA(0));
  asyncB(0, 0);
  asm volatile("s_wait_asynccnt 0x0" ::: "memory");
  __syncthreads();

  union Frag { uint4 u[2]; v16h h; };
  for (int ch = 0; ch < TOT_CH; ++ch) {
    const int cur = ch & 1;
    const _Float16* cA = bufA(cur);
    const _Float16* cB = bufB(cur);
    // ---- A fragments from current buffer (ISA layout: lanes<16 K{0..7,16..23})
    Frag a[2];
    {
      const _Float16* p0 = cA + aFragOff0;
      a[0].u[0] = *reinterpret_cast<const uint4*>(p0);
      a[0].u[1] = *reinterpret_cast<const uint4*>(p0 + 16);
      const _Float16* p1 = cA + aFragOff1;
      a[1].u[0] = *reinterpret_cast<const uint4*>(p1);
      a[1].u[1] = *reinterpret_cast<const uint4*>(p1 + 16);
    }
    // ---- issue next-chunk staging now; WMMAs below hide the latency.
    //      B goes straight to LDS via async DMA (zero data registers).
    uint4 aS = {0u, 0u, 0u, 0u};
    const bool more = (ch + 1 < TOT_CH);
    if (more) { aS = loadA(ch + 1); asyncB(ch + 1, cur ^ 1); }
    // ---- matrix work: B fragments in two halves to cap register pressure
#pragma unroll
    for (int half = 0; half < 2; ++half) {
      Frag bf[2];
#pragma unroll
      for (int nf = 0; nf < 2; ++nf) {
        const _Float16* p = cB + bFragBase + (half * 2 + nf) * 16 * B_STRIDE;
        bf[nf].u[0] = *reinterpret_cast<const uint4*>(p);
        bf[nf].u[1] = *reinterpret_cast<const uint4*>(p + 16);
      }
#pragma unroll
      for (int mf = 0; mf < 2; ++mf)
#pragma unroll
        for (int nf = 0; nf < 2; ++nf)
          acc[mf][half * 2 + nf] = __builtin_amdgcn_wmma_f32_16x16x32_f16(
              false, a[mf].h, false, bf[nf].h, (short)0, acc[mf][half * 2 + nf],
              false, false);
      // keep the two halves' ds_loads from being merged (register pressure)
      __builtin_amdgcn_sched_barrier(0);
    }
    // ---- stage next A tile, drain async B copies, publish buffer
    if (more) storeA(cur ^ 1, aS);
    asm volatile("s_wait_asynccnt 0x0" ::: "memory");
    __syncthreads();   // reads of buf[cur] done; next iter may overwrite it
  }

  // ---- exchange accumulators through LDS, fuse gates
#pragma unroll
  for (int mf = 0; mf < 2; ++mf)
#pragma unroll
    for (int nf = 0; nf < 4; ++nf) {
      const int Nl = wn * 64 + nf * 16 + l16;
      const int Mb = wm * 32 + mf * 16 + (hsel ? 8 : 0);
#pragma unroll
      for (int i = 0; i < 8; ++i)
        ldsC[(Mb + i) * NGATE + Nl] = acc[mf][nf][i];
    }
  __syncthreads();

#pragma unroll 4
  for (int it = 0; it < (BLK_M * FF) / NTH; ++it) {
    const int idx = tid + it * NTH;           // 0..4095
    const int m = idx >> 6, j = idx & (FF - 1);
    const int gmj = (gm0 + m) * FF + j;
    const float zi = ldsC[m * NGATE + j]            + bias[j];
    const float zf = ldsC[m * NGATE + FF + j]       + bias[FF + j];
    const float zc = ldsC[m * NGATE + 2 * FF + j]   + bias[2 * FF + j];
    const float zo = ldsC[m * NGATE + 3 * FF + j]   + bias[3 * FF + j];
    const float ig = hsig(zi), fg = hsig(zf), og = hsig(zo);
    const float cp = HAS_STATE ? cstate[gmj] : 0.0f;
    const float cn = fg * cp + ig * tanhf(zc);
    cstate[gmj] = cn;
    hout[gmj]   = (_Float16)(og * tanhf(cn));
  }
}

// ---------------------------------------------------------------------------
// Support kernels: cast/pack/BN/out-conv
// ---------------------------------------------------------------------------
__global__ void cast_x_kernel(const float* __restrict__ x, _Float16* __restrict__ x16) {
  const int i = blockIdx.x * blockDim.x + threadIdx.x;   // t-major pixel index
  if (i >= TT * BB * HW) return;
  const int p = i % HW;
  const int b = (i / HW) % BB;
  const int t = i / (HW * BB);
  const float* src = x + ((size_t)(b * TT + t) * HW + p) * CC;
  _Float16* dst = x16 + (size_t)i * 4;
  dst[0] = (_Float16)src[0];
  dst[1] = (_Float16)src[1];
  dst[2] = (_Float16)src[2];
  dst[3] = (_Float16)0.0f;                               // zero pad channel
}

__global__ void pack_w_c64_kernel(const float* __restrict__ w, _Float16* __restrict__ out) {
  const int i = blockIdx.x * blockDim.x + threadIdx.x;   // 18*256*32
  if (i >= KCH_H * NGATE * 32) return;
  const int kk = i & 31;
  const int n  = (i >> 5) & 255;
  const int c  = i >> 13;
  const int tap = c >> 1;
  const int ci  = (c & 1) * 32 + kk;
  out[i] = (_Float16)w[((size_t)tap * FF + ci) * NGATE + n];
}

__global__ void pack_w_c4_kernel(const float* __restrict__ w, _Float16* __restrict__ out) {
  const int i = blockIdx.x * blockDim.x + threadIdx.x;   // 2*256*32
  if (i >= 2 * NGATE * 32) return;
  const int kk = i & 31;
  const int n  = (i >> 5) & 255;
  const int c  = i >> 13;
  float v = 0.0f;
  if (c == 0) {
    const int tap = kk >> 2, ci = kk & 3;
    if (ci < 3) v = w[((size_t)tap * CC + ci) * NGATE + n];
  } else if (kk < 3) {
    v = w[((size_t)8 * CC + kk) * NGATE + n];
  }
  out[i] = (_Float16)v;
}

__global__ void bn_cast_kernel(const _Float16* __restrict__ h, const float* __restrict__ g,
                               const float* __restrict__ be, const float* __restrict__ mn,
                               const float* __restrict__ vr, _Float16* __restrict__ out, int total) {
  const int i = blockIdx.x * blockDim.x + threadIdx.x;
  if (i >= total) return;
  const int j = i & (FF - 1);
  const float s = g[j] * rsqrtf(vr[j] + 1e-3f);
  out[i] = (_Float16)(((float)h[i] - mn[j]) * s + be[j]);
}

__global__ __launch_bounds__(256) void out_conv_kernel(const _Float16* __restrict__ d,
                                                       const float* __restrict__ w,
                                                       const float* __restrict__ ob,
                                                       float* __restrict__ y) {
  __shared__ float wsW[27 * FF * CC];                    // 20.25 KB, reused by all threads
  for (int i = threadIdx.x; i < 27 * FF * CC; i += 256) wsW[i] = w[i];
  __syncthreads();
  const int gid = blockIdx.x * 256 + threadIdx.x;        // (b*T + t)*HW + p, output order
  if (gid >= BB * TT * HW) return;
  const int p = gid % HW;
  const int t = (gid / HW) % TT;
  const int b = gid / (HW * TT);
  const int py0 = p / WW, px0 = p % WW;
  float a0 = ob[0], a1 = ob[1], a2 = ob[2];
  for (int dt = 0; dt < 3; ++dt) {
    const int ts = t + dt - 1;
    if ((unsigned)ts >= (unsigned)TT) continue;
    for (int dy = 0; dy < 3; ++dy) {
      const int py = py0 + dy - 1;
      if ((unsigned)py >= (unsigned)HH) continue;
      for (int dx = 0; dx < 3; ++dx) {
        const int px = px0 + dx - 1;
        if ((unsigned)px >= (unsigned)WW) continue;
        const _Float16* s = d + ((size_t)(ts * BB + b) * HW + py * WW + px) * FF;
        const float* wp = wsW + ((dt * 9 + dy * 3 + dx) * FF) * CC;
#pragma unroll 4
        for (int ci = 0; ci < FF; ++ci) {
          const float v = (float)s[ci];
          a0 += v * wp[ci * 3 + 0];
          a1 += v * wp[ci * 3 + 1];
          a2 += v * wp[ci * 3 + 2];
        }
      }
    }
  }
  const size_t o = (size_t)gid * 3;
  y[o + 0] = 1.0f / (1.0f + expf(-a0));
  y[o + 1] = 1.0f / (1.0f + expf(-a1));
  y[o + 2] = 1.0f / (1.0f + expf(-a2));
}

// ---------------------------------------------------------------------------
extern "C" void kernel_launch(void* const* d_in, const int* in_sizes, int n_in,
                              void* d_out, int out_size, void* d_ws, size_t ws_size,
                              hipStream_t stream) {
  const float* x      = (const float*)d_in[0];
  const float* enc_Wx = (const float*)d_in[1];
  const float* enc_Wh = (const float*)d_in[2];
  const float* enc_b  = (const float*)d_in[3];
  const float* enc_g  = (const float*)d_in[4];
  const float* enc_be = (const float*)d_in[5];
  const float* enc_mn = (const float*)d_in[6];
  const float* enc_vr = (const float*)d_in[7];
  const float* dec_Wx = (const float*)d_in[8];
  const float* dec_Wh = (const float*)d_in[9];
  const float* dec_b  = (const float*)d_in[10];
  const float* dec_g  = (const float*)d_in[11];
  const float* dec_be = (const float*)d_in[12];
  const float* dec_mn = (const float*)d_in[13];
  const float* dec_vr = (const float*)d_in[14];
  const float* out_W  = (const float*)d_in[15];
  const float* out_b  = (const float*)d_in[16];

  char* cur = (char*)d_ws;
  auto take = [&](size_t bytes) -> char* {
    char* p = cur;
    cur += (bytes + 255) & ~(size_t)255;
    return p;
  };

  const size_t sliceX = (size_t)BB * HW * 4;   // halfs per timestep (padded x)
  const size_t sliceF = (size_t)BB * HW * FF;  // halfs per timestep (feature maps)

  _Float16* x16    = (_Float16*)take(TT * sliceX * 2);
  _Float16* hEnc   = (_Float16*)take(TT * sliceF * 2);
  _Float16* eBuf   = (_Float16*)take(TT * sliceF * 2);
  _Float16* hDec   = (_Float16*)take(TT * sliceF * 2);
  _Float16* dBuf   = (_Float16*)take(TT * sliceF * 2);
  float*    cEnc   = (float*)take(sliceF * 4);
  float*    cDec   = (float*)take(sliceF * 4);
  _Float16* wxEncP = (_Float16*)take((size_t)2 * NGATE * 32 * 2);
  _Float16* whEncP = (_Float16*)take((size_t)KCH_H * NGATE * 32 * 2);
  _Float16* wxDecP = (_Float16*)take((size_t)KCH_H * NGATE * 32 * 2);
  _Float16* whDecP = (_Float16*)take((size_t)KCH_H * NGATE * 32 * 2);

  // One-time packing / casting (deterministic, every call)
  pack_w_c4_kernel<<<(2 * NGATE * 32 + 255) / 256, 256, 0, stream>>>(enc_Wx, wxEncP);
  pack_w_c64_kernel<<<(KCH_H * NGATE * 32 + 255) / 256, 256, 0, stream>>>(enc_Wh, whEncP);
  pack_w_c64_kernel<<<(KCH_H * NGATE * 32 + 255) / 256, 256, 0, stream>>>(dec_Wx, wxDecP);
  pack_w_c64_kernel<<<(KCH_H * NGATE * 32 + 255) / 256, 256, 0, stream>>>(dec_Wh, whDecP);
  cast_x_kernel<<<(TT * BB * HW + 255) / 256, 256, 0, stream>>>(x, x16);

  const int nblk = MTOT / BLK_M;  // 2048 workgroups per step
  const int tot  = TT * BB * HW * FF;

  // Encoder ConvLSTM (sequential in t; each step is a WMMA implicit GEMM)
  lstm_step_kernel<true, false><<<nblk, NTH, 0, stream>>>(
      x16, nullptr, wxEncP, whEncP, enc_b, hEnc, cEnc);
  for (int t = 1; t < TT; ++t)
    lstm_step_kernel<true, true><<<nblk, NTH, 0, stream>>>(
        x16 + (size_t)t * sliceX, hEnc + (size_t)(t - 1) * sliceF,
        wxEncP, whEncP, enc_b, hEnc + (size_t)t * sliceF, cEnc);
  bn_cast_kernel<<<(tot + 255) / 256, 256, 0, stream>>>(hEnc, enc_g, enc_be, enc_mn, enc_vr, eBuf, tot);

  // Decoder ConvLSTM
  lstm_step_kernel<false, false><<<nblk, NTH, 0, stream>>>(
      eBuf, nullptr, wxDecP, whDecP, dec_b, hDec, cDec);
  for (int t = 1; t < TT; ++t)
    lstm_step_kernel<false, true><<<nblk, NTH, 0, stream>>>(
        eBuf + (size_t)t * sliceF, hDec + (size_t)(t - 1) * sliceF,
        wxDecP, whDecP, dec_b, hDec + (size_t)t * sliceF, cDec);
  bn_cast_kernel<<<(tot + 255) / 256, 256, 0, stream>>>(hDec, dec_g, dec_be, dec_mn, dec_vr, dBuf, tot);

  // 3D output conv + sigmoid (NUM_PRED == T, so no slicing needed)
  out_conv_kernel<<<(BB * TT * HW) / 256, 256, 0, stream>>>(dBuf, out_W, out_b, (float*)d_out);

  (void)in_sizes; (void)n_in; (void)out_size; (void)ws_size;
}
